// IpaMultiRigidDenoiser_31937376813592
// MI455X (gfx1250) — matrix operand order repaired and checked by reading, not compiled, not run on hardware
//
#include <hip/hip_runtime.h>
#include <hip/hip_bf16.h>
#include <math.h>

// ---------------- problem constants ----------------
#define BB 2
#define NN 512
#define CS 384
#define CZ 128
#define HH 4
#define CC 32
#define PQK 4
#define PV 8
#define IE 256
#define NAA 21
#define NRBF 16
#define NREL 65

#define ROWS (BB*NN)          // 1024
#define LINC 576              // 3*128 + 2*48 + 96
#define FEAT 768              // H*(C+CZ+PV*4)
#define KATT 64               // packed score K (32 ch + 12 pts + pad)

typedef __bf16 bf16_t;
typedef __attribute__((ext_vector_type(16))) __bf16 bf16x16;
typedef __attribute__((ext_vector_type(8)))  float   f32x8;

__device__ __forceinline__ bf16_t f2bf(float x){ return (bf16_t)x; }

// WL = sqrt(1/3), WC = sqrt(2/(9*PQK))
#define WLs 0.57735026918962576f
#define WCs 0.23570226039551584f

// ---------------- WMMA fragment helpers ----------------
// A (16x32 bf16): lane m = lane&15, group g = lane>>4.
//   elems [0..15] = A[m][k0 + g*16 + 0..15]
__device__ __forceinline__ bf16x16 load_a_frag(const bf16_t* rowp, int k0, int g){
    bf16x16 f;
    ((uint4*)&f)[0] = *(const uint4*)(rowp + k0 + g*16);
    ((uint4*)&f)[1] = *(const uint4*)(rowp + k0 + g*16 + 8);
    return f;
}
// B (32x16 bf16) from K-major storage Bt[n][k]: lane n = lane&15, group g.
//   elems [0..15] = Bt[n][k0 + g*16 + 0..15]   (one contiguous 32B run)
__device__ __forceinline__ bf16x16 load_b_frag(const bf16_t* rowp, int k0, int g){
    bf16x16 f;
    const bf16_t* p = rowp + k0 + g*16;
    ((uint4*)&f)[0] = *(const uint4*)(p);
    ((uint4*)&f)[1] = *(const uint4*)(p + 8);
    return f;
}
__device__ __forceinline__ f32x8 wmma_bf16(bf16x16 a, bf16x16 b, f32x8 c){
    return __builtin_amdgcn_wmma_f32_16x16x32_bf16(false, a, false, b, (short)0, c, false, false);
}
__device__ __forceinline__ unsigned lds_off32(const void* p){
    // generic LDS address: low 32 bits are the wave-relative LDS byte offset
    return (unsigned)(unsigned long long)p;
}

// ---------------- simple 1-wave WMMA GEMM (narrow N, e.g. P@V) ----------------
// C[b][m][n] = sum_k A[b][m][k] * Bt[b][n][k]   (+ Cadd)
// grid: (Nc/16, M/16, batch), block: 32
__global__ void k_wmma_gemm(const bf16_t* __restrict__ A, const bf16_t* __restrict__ Bt,
                            float* __restrict__ Cout, const float* __restrict__ Cadd,
                            int K, int lda, int ldb, int ldc,
                            long long sA, long long sB, long long sC)
{
    int lane = threadIdx.x & 31;
    int g = lane >> 4, r = lane & 15;
    const bf16_t* Ap = A  + (long long)blockIdx.z*sA + (long long)(blockIdx.y*16 + r)*lda;
    const bf16_t* Bp = Bt + (long long)blockIdx.z*sB + (long long)(blockIdx.x*16 + r)*ldb;
    f32x8 acc = {0.f,0.f,0.f,0.f,0.f,0.f,0.f,0.f};
    for (int k0 = 0; k0 < K; k0 += 32){
        if (k0 + 64 <= K){ __builtin_prefetch(Ap + k0 + 64, 0, 1); __builtin_prefetch(Bp + k0 + 64, 0, 1); }
        acc = wmma_bf16(load_a_frag(Ap, k0, g), load_b_frag(Bp, k0, g), acc);
    }
    int n = blockIdx.x*16 + r;
    long long base = (long long)blockIdx.z*sC + (long long)(blockIdx.y*16 + g*8)*ldc + n;
#pragma unroll
    for (int i = 0; i < 8; i++){
        float v = acc[i];
        if (Cadd) v += Cadd[base + (long long)i*ldc];
        Cout[base + (long long)i*ldc] = v;
    }
}

// ---------------- 4-wave 64x64 block WMMA GEMM with async-LDS B staging ----------
// Each wave computes a 16x64 strip (4 accumulators, A fragment reused 4x).
// B panel (64 x 32 bf16 = 4KB) staged via global_load_async_to_lds_b128
// (ASYNCcnt) with double buffering; fragments read back via ds_load_b128.
// grid: (Nc/64, M/64), block: 128
__global__ void k_wmma_gemm_blk(const bf16_t* __restrict__ A, const bf16_t* __restrict__ Bt,
                                float* __restrict__ Cout, const float* __restrict__ Cadd,
                                int K, int lda, int ldb, int ldc)
{
    __shared__ bf16_t ldsB[2][64*32];
    int tid  = threadIdx.x;           // 128
    int wave = tid >> 5, lane = tid & 31;
    int g = lane >> 4, r = lane & 15;
    int m0 = blockIdx.y*64 + wave*16;
    int n0 = blockIdx.x*64;
    const bf16_t* Ap    = A  + (long long)(m0 + r)*lda;
    const bf16_t* Bbase = Bt + (long long)n0*ldb;

    f32x8 zero = {0.f,0.f,0.f,0.f,0.f,0.f,0.f,0.f};
    f32x8 acc[4];
#pragma unroll
    for (int t = 0; t < 4; t++) acc[t] = zero;

    const int S = K >> 5;             // k-steps of 32
    // stage step 0 into buffer 0: 64x32 tile, 16B per thread per issue
    {
        unsigned b0 = lds_off32(&ldsB[0][0]);
#pragma unroll
        for (int q = 0; q < 2; q++){
            int e  = tid*8 + q*1024;          // bf16 element index in tile
            int n  = e >> 5, kk = e & 31;
            const bf16_t* gp = Bbase + (long long)n*ldb + kk;
            unsigned dst = b0 + (unsigned)(e*2);
            asm volatile("global_load_async_to_lds_b128 %0, %1, off"
                         :: "v"(dst), "v"((unsigned long long)gp) : "memory");
        }
    }
    for (int s = 0; s < S; s++){
        asm volatile("s_wait_asynccnt 0" ::: "memory");
        __syncthreads();
        if (s + 1 < S){
            unsigned bn = lds_off32(&ldsB[(s+1)&1][0]);
            int k0n = (s+1)*32;
#pragma unroll
            for (int q = 0; q < 2; q++){
                int e  = tid*8 + q*1024;
                int n  = e >> 5, kk = e & 31;
                const bf16_t* gp = Bbase + (long long)n*ldb + k0n + kk;
                unsigned dst = bn + (unsigned)(e*2);
                asm volatile("global_load_async_to_lds_b128 %0, %1, off"
                             :: "v"(dst), "v"((unsigned long long)gp) : "memory");
            }
        }
        bf16x16 af = load_a_frag(Ap, s*32, g);
        const bf16_t* lb = &ldsB[s&1][0];
#pragma unroll
        for (int t = 0; t < 4; t++){
            bf16x16 bfr;
            const bf16_t* pp = lb + ((t*16 + r)*32 + g*16);
            ((uint4*)&bfr)[0] = *(const uint4*)(pp);      // ds_load_b128
            ((uint4*)&bfr)[1] = *(const uint4*)(pp + 8);  // ds_load_b128
            acc[t] = wmma_bf16(af, bfr, acc[t]);
        }
        __syncthreads();
    }
#pragma unroll
    for (int t = 0; t < 4; t++){
        int n = n0 + t*16 + r;
        long long base = (long long)(m0 + g*8)*ldc + n;
#pragma unroll
        for (int i = 0; i < 8; i++){
            float v = acc[t][i];
            if (Cadd) v += Cadd[base + (long long)i*ldc];
            Cout[base + (long long)i*ldc] = v;
        }
    }
}

// ---------------- small precompute (1 block) ----------------
__global__ void k_prep(const float* __restrict__ t, const float* __restrict__ gamma,
                       const float* __restrict__ W_time, const float* __restrict__ W_relpos,
                       const float* __restrict__ W_rbf, const float* __restrict__ W_chain,
                       const float* __restrict__ W_bond, const float* __restrict__ Wb,
                       float* tvec, float* WbRel, float* WbRbf, float* WbCh, float* WbBd, float* cgam)
{
    int tid = threadIdx.x; // 256
    if (tid < HH){
        float gm = gamma[tid];
        float sp = (gm > 20.f) ? gm : log1pf(__expf(gm));
        cgam[tid] = WLs * WCs * 0.5f * sp;
        float ch = 0.f, bd = 0.f;
        for (int z = 0; z < CZ; z++){ ch += W_chain[z]*Wb[z*HH+tid]; bd += W_bond[z]*Wb[z*HH+tid]; }
        WbCh[tid] = ch; WbBd[tid] = bd;
    }
    for (int idx = tid; idx < NREL*HH; idx += 256){
        int rel = idx >> 2, h = idx & 3; float s = 0.f;
        for (int z = 0; z < CZ; z++) s += W_relpos[rel*CZ+z]*Wb[z*HH+h];
        WbRel[idx] = s;
    }
    for (int idx = tid; idx < NRBF*HH; idx += 256){
        int rr = idx >> 2, h = idx & 3; float s = 0.f;
        for (int z = 0; z < CZ; z++) s += W_rbf[rr*CZ+z]*Wb[z*HH+h];
        WbRbf[idx] = s;
    }
    const float lg = logf(10000.f) / 127.f;
    for (int idx = tid; idx < BB*CS; idx += 256){
        int b = idx / CS, c = idx % CS;
        float tv = t[b]*10000.f, s = 0.f;
        for (int k = 0; k < IE/2; k++){
            float ang = tv * __expf(-(float)k * lg);
            s += __sinf(ang)*W_time[k*CS+c] + __cosf(ang)*W_time[(IE/2+k)*CS+c];
        }
        tvec[idx] = s;
    }
}

// ---------------- weight packing (bf16, K-major) ----------------
__global__ void k_pack_weights(const float* __restrict__ Wq, const float* __restrict__ Wk,
                               const float* __restrict__ Wv, const float* __restrict__ Wqp,
                               const float* __restrict__ Wkp, const float* __restrict__ Wvp,
                               const float* __restrict__ Wout,
                               bf16_t* __restrict__ WcatT, bf16_t* __restrict__ WoutT)
{
    const int T1 = LINC*CS, T2 = CS*FEAT;
    for (int idx = blockIdx.x*blockDim.x + threadIdx.x; idx < T1+T2; idx += gridDim.x*blockDim.x){
        if (idx < T1){
            int rr = idx / CS, k = idx % CS; float v;
            if      (rr < 128) v = Wq [k*128 + rr];
            else if (rr < 256) v = Wk [k*128 + rr-128];
            else if (rr < 384) v = Wv [k*128 + rr-256];
            else if (rr < 432) v = Wqp[k*48  + rr-384];
            else if (rr < 480) v = Wkp[k*48  + rr-432];
            else               v = Wvp[k*96  + rr-480];
            WcatT[idx] = f2bf(v);
        } else {
            int j = idx - T1, rr = j / FEAT, k = j % FEAT;
            WoutT[j] = f2bf(Wout[k*CS + rr]);
        }
    }
}

// ---------------- node embedding + rigid frames ----------------
__global__ void k_node(const int* __restrict__ seq, const int* __restrict__ seq_idx,
                       const float* __restrict__ rigid7, const float* __restrict__ W_node_idx,
                       const float* __restrict__ W_seq, const float* __restrict__ tvec,
                       float* __restrict__ node, bf16_t* __restrict__ nodeBF,
                       float* __restrict__ Rm, float* __restrict__ trans)
{
    __shared__ float ie[IE];
    int row = blockIdx.x;              // b*N + n
    int b = row / NN;
    int tid = threadIdx.x;             // 384
    int idx = seq_idx[row];
    if (tid < IE/2){
        float ang = (float)idx * 3.14159265358979323846f * __powf(2056.f, -2.f*(float)tid/(float)IE);
        ie[tid] = __sinf(ang); ie[tid+IE/2] = __cosf(ang);
    }
    if (tid == 0){
        const float* q7 = rigid7 + row*7;
        float qw=q7[0], qx=q7[1], qy=q7[2], qz=q7[3];
        float inv = rsqrtf(qw*qw+qx*qx+qy*qy+qz*qz + 1e-8f);
        qw*=inv; qx*=inv; qy*=inv; qz*=inv;
        float* Rr = Rm + row*9;
        Rr[0]=1.f-2.f*(qy*qy+qz*qz); Rr[1]=2.f*(qx*qy-qw*qz);     Rr[2]=2.f*(qx*qz+qw*qy);
        Rr[3]=2.f*(qx*qy+qw*qz);     Rr[4]=1.f-2.f*(qx*qx+qz*qz); Rr[5]=2.f*(qy*qz-qw*qx);
        Rr[6]=2.f*(qx*qz-qw*qy);     Rr[7]=2.f*(qy*qz+qw*qx);     Rr[8]=1.f-2.f*(qx*qx+qy*qy);
        trans[row*4+0]=q7[4]; trans[row*4+1]=q7[5]; trans[row*4+2]=q7[6]; trans[row*4+3]=0.f;
    }
    __syncthreads();
    int aa = seq[row];
    float acc = W_seq[aa*CS + tid] + tvec[b*CS + tid];
    for (int k = 0; k < IE; k++) acc += ie[k] * W_node_idx[k*CS + tid];
    node[row*CS + tid] = acc;
    nodeBF[row*CS + tid] = f2bf(acc);
}

// ---------------- pack q/k/v + rigid-transformed points ----------------
__global__ void k_pack_qkv(const float* __restrict__ lin, const float* __restrict__ Rm,
                           const float* __restrict__ trans, const float* __restrict__ cgam,
                           bf16_t* __restrict__ Aq, bf16_t* __restrict__ Ak,
                           float* __restrict__ dk, bf16_t* __restrict__ vT, bf16_t* __restrict__ vgT)
{
    for (int idx = blockIdx.x*blockDim.x + threadIdx.x; idx < ROWS*HH; idx += gridDim.x*blockDim.x){
        int h = idx & 3, row = idx >> 2;      // row = b*N+n
        int b = row / NN, n = row % NN;
        int z = b*HH + h;
        const float* L = lin + (long long)row*LINC;
        const float* Rr = Rm + row*9;
        const float* tr = trans + row*4;
        float ch = cgam[h];
        float qs = WLs * rsqrtf((float)CC);

        bf16_t* aq = Aq + ((long long)z*NN + n)*KATT;
        bf16_t* ak = Ak + ((long long)z*NN + n)*KATT;
        for (int c = 0; c < CC; c++){
            aq[c] = f2bf(L[h*CC + c] * qs);
            ak[c] = f2bf(L[128 + h*CC + c]);
        }
        float kn2 = 0.f;
        for (int p = 0; p < PQK; p++){
            const float* qp = L + 384 + h*12 + p*3;
            const float* kp = L + 432 + h*12 + p*3;
            for (int d = 0; d < 3; d++){
                float qg = Rr[d*3+0]*qp[0] + Rr[d*3+1]*qp[1] + Rr[d*3+2]*qp[2] + tr[d];
                float kg = Rr[d*3+0]*kp[0] + Rr[d*3+1]*kp[1] + Rr[d*3+2]*kp[2] + tr[d];
                aq[CC + p*3 + d] = f2bf(2.f*ch*qg);
                ak[CC + p*3 + d] = f2bf(kg);
                kn2 += kg*kg;
            }
        }
        for (int e = CC+12; e < KATT; e++){ aq[e] = f2bf(0.f); ak[e] = f2bf(0.f); }
        dk[(long long)z*NN + n] = ch * kn2;

        for (int c = 0; c < CC; c++)
            vT[((long long)z*CC + c)*NN + n] = f2bf(L[256 + h*CC + c]);
        for (int p = 0; p < PV; p++){
            const float* vp = L + 480 + h*24 + p*3;
            for (int d = 0; d < 3; d++){
                float vg = Rr[d*3+0]*vp[0] + Rr[d*3+1]*vp[1] + Rr[d*3+2]*vp[2] + tr[d];
                vgT[((long long)z*32 + p*3 + d)*NN + n] = f2bf(vg);
            }
        }
        for (int rr = 24; rr < 32; rr++) vgT[((long long)z*32 + rr)*NN + n] = f2bf(0.f);
    }
}

// ---------------- pair bias (projected through Wb, z never materialized) ----------------
__global__ void k_pair(const int* __restrict__ seq_idx, const int* __restrict__ chain_idx,
                       const float* __restrict__ bonds, const float* __restrict__ trans,
                       const float* __restrict__ WbRel, const float* __restrict__ WbRbf,
                       const float* __restrict__ WbCh, const float* __restrict__ WbBd,
                       float* __restrict__ wlbias)
{
    const float step = 20.f/(NRBF-1);
    const float inv2s2 = 1.f / (2.f * (20.f/NRBF) * (20.f/NRBF));
    for (long long idx = blockIdx.x*(long long)blockDim.x + threadIdx.x; idx < (long long)BB*NN*NN;
         idx += (long long)gridDim.x*blockDim.x){
        int j = (int)(idx % NN);
        int i = (int)((idx / NN) % NN);
        int b = (int)(idx / (NN*NN));
        const float* ti = trans + (b*NN+i)*4;
        const float* tj = trans + (b*NN+j)*4;
        float dx=ti[0]-tj[0], dy=ti[1]-tj[1], dz=ti[2]-tj[2];
        float d = sqrtf(dx*dx+dy*dy+dz*dz + 1e-8f);
        int rel = seq_idx[b*NN+i] - seq_idx[b*NN+j];
        rel = (rel < -32) ? -32 : (rel > 32 ? 32 : rel); rel += 32;
        float eq = (chain_idx[b*NN+i] == chain_idx[b*NN+j]) ? 1.f : 0.f;
        float bd = bonds[idx];
        float bh[HH];
#pragma unroll
        for (int h = 0; h < HH; h++) bh[h] = WbRel[rel*HH+h] + eq*WbCh[h] + bd*WbBd[h];
        for (int r = 0; r < NRBF; r++){
            float dd = d - (float)r*step;
            float w = __expf(-dd*dd*inv2s2);
#pragma unroll
            for (int h = 0; h < HH; h++) bh[h] += w*WbRbf[r*HH+h];
        }
        float* o = wlbias + idx*HH;
#pragma unroll
        for (int h = 0; h < HH; h++) o[h] = WLs * bh[h];
    }
}

// ---------------- attention scores (WMMA, K=64 packed) ----------------
// grid: (N/16 j, N/16 i, B*H), block 32
__global__ void k_scores(const bf16_t* __restrict__ Aq, const bf16_t* __restrict__ Ak,
                         const float* __restrict__ wlbias, const float* __restrict__ dk,
                         const float* __restrict__ mask, float* __restrict__ logits)
{
    int lane = threadIdx.x & 31;
    int g = lane >> 4, r = lane & 15;
    int z = blockIdx.z, b = z / HH, h = z % HH;
    const bf16_t* Ap = Aq + ((long long)z*NN + blockIdx.y*16 + r)*KATT;
    const bf16_t* Bp = Ak + ((long long)z*NN + blockIdx.x*16 + r)*KATT;
    f32x8 acc = {0.f,0.f,0.f,0.f,0.f,0.f,0.f,0.f};
    acc = wmma_bf16(load_a_frag(Ap,  0, g), load_b_frag(Bp,  0, g), acc);
    acc = wmma_bf16(load_a_frag(Ap, 32, g), load_b_frag(Bp, 32, g), acc);
    int j = blockIdx.x*16 + r;
    float mterm = (mask[b*NN + j] - 1.f) * 1e9f;
    float dkj = dk[(long long)z*NN + j];
#pragma unroll
    for (int q = 0; q < 8; q++){
        int i = blockIdx.y*16 + g*8 + q;
        float v = acc[q] + wlbias[(((long long)b*NN + i)*NN + j)*HH + h] - dkj + mterm;
        logits[((long long)z*NN + i)*NN + j] = v;
    }
}

// ---------------- softmax (one block per row) ----------------
__global__ void k_softmax(const float* __restrict__ logits, bf16_t* __restrict__ probsBF)
{
    __shared__ float red[128];
    long long row = blockIdx.x;
    const float* L = logits + row*NN;
    int tid = threadIdx.x;
    float v[4];
    float m = -3.402823e38f;
#pragma unroll
    for (int q = 0; q < 4; q++){ v[q] = L[tid + 128*q]; m = fmaxf(m, v[q]); }
    red[tid] = m; __syncthreads();
    for (int s = 64; s > 0; s >>= 1){ if (tid < s) red[tid] = fmaxf(red[tid], red[tid+s]); __syncthreads(); }
    m = red[0]; __syncthreads();
    float l = 0.f;
#pragma unroll
    for (int q = 0; q < 4; q++){ v[q] = __expf(v[q] - m); l += v[q]; }
    red[tid] = l; __syncthreads();
    for (int s = 64; s > 0; s >>= 1){ if (tid < s) red[tid] += red[tid+s]; __syncthreads(); }
    float inv = 1.f / red[0];
#pragma unroll
    for (int q = 0; q < 4; q++) probsBF[row*NN + tid + 128*q] = f2bf(v[q] * inv);
}

// ---------------- opair via factored statistics ----------------
__global__ void k_opair(const bf16_t* __restrict__ probsBF, const int* __restrict__ seq_idx,
                        const int* __restrict__ chain_idx, const float* __restrict__ bonds,
                        const float* __restrict__ trans, const float* __restrict__ W_relpos,
                        const float* __restrict__ W_rbf, const float* __restrict__ W_chain,
                        const float* __restrict__ W_bond, bf16_t* __restrict__ featsBF)
{
    __shared__ float sRel[HH*NREL];
    __shared__ float sRbf[HH*NRBF];
    __shared__ float sCh[HH], sBd[HH];
    int tid = threadIdx.x;                 // 128
    int row = blockIdx.x;                  // b*N + i
    int b = row / NN, i = row % NN;
    for (int q = tid; q < HH*NREL; q += 128) sRel[q] = 0.f;
    for (int q = tid; q < HH*NRBF; q += 128) sRbf[q] = 0.f;
    if (tid < HH){ sCh[tid] = 0.f; sBd[tid] = 0.f; }
    __syncthreads();

    const float step = 20.f/(NRBF-1);
    const float inv2s2 = 1.f / (2.f * (20.f/NRBF) * (20.f/NRBF));
    const float* ti = trans + row*4;
    int si = seq_idx[row], ci = chain_idx[row];
    for (int j = tid; j < NN; j += 128){
        float a[HH];
#pragma unroll
        for (int h = 0; h < HH; h++)
            a[h] = (float)probsBF[(((long long)(b*HH+h))*NN + i)*NN + j];
        int rel = si - seq_idx[b*NN+j];
        rel = (rel < -32) ? -32 : (rel > 32 ? 32 : rel); rel += 32;
        float eq = (ci == chain_idx[b*NN+j]) ? 1.f : 0.f;
        float bd = bonds[((long long)b*NN + i)*NN + j];
        const float* tj = trans + (b*NN+j)*4;
        float dx=ti[0]-tj[0], dy=ti[1]-tj[1], dz=ti[2]-tj[2];
        float d = sqrtf(dx*dx+dy*dy+dz*dz + 1e-8f);
#pragma unroll
        for (int h = 0; h < HH; h++){
            atomicAdd(&sRel[h*NREL + rel], a[h]);
            if (eq != 0.f) atomicAdd(&sCh[h], a[h]);
            if (bd != 0.f) atomicAdd(&sBd[h], a[h]*bd);
        }
        for (int r = 0; r < NRBF; r++){
            float dd = d - (float)r*step;
            float w = __expf(-dd*dd*inv2s2);
#pragma unroll
            for (int h = 0; h < HH; h++) atomicAdd(&sRbf[h*NRBF + r], a[h]*w);
        }
    }
    __syncthreads();
    for (int q = tid; q < HH*CZ; q += 128){
        int h = q / CZ, cz = q % CZ;
        float s = sCh[h]*W_chain[cz] + sBd[h]*W_bond[cz];
        for (int rel = 0; rel < NREL; rel++) s += sRel[h*NREL+rel]*W_relpos[rel*CZ+cz];
        for (int r = 0; r < NRBF; r++)      s += sRbf[h*NRBF+r]*W_rbf[r*CZ+cz];
        featsBF[(long long)row*FEAT + 256 + q] = f2bf(s);
    }
}

// ---------------- finalize o / ol / onorm into feats ----------------
__global__ void k_finalize(const float* __restrict__ obuf, const float* __restrict__ ogbuf,
                           const float* __restrict__ Rm, const float* __restrict__ trans,
                           bf16_t* __restrict__ featsBF)
{
    for (int idx = blockIdx.x*blockDim.x + threadIdx.x; idx < ROWS*HH; idx += gridDim.x*blockDim.x){
        int h = idx & 3, row = idx >> 2;
        int b = row / NN, n = row % NN;
        int z = b*HH + h;
        const float* ob = obuf  + ((long long)z*NN + n)*32;
        const float* og = ogbuf + ((long long)z*NN + n)*32;
        const float* Rr = Rm + row*9;
        const float* tr = trans + row*4;
        bf16_t* f = featsBF + (long long)row*FEAT;
        for (int c = 0; c < CC; c++) f[h*CC + c] = f2bf(ob[c]);
        for (int p = 0; p < PV; p++){
            float vx = og[p*3+0]-tr[0], vy = og[p*3+1]-tr[1], vz = og[p*3+2]-tr[2];
            float ol[3]; float nrm = 1e-8f;
#pragma unroll
            for (int d = 0; d < 3; d++){
                ol[d] = Rr[0*3+d]*vx + Rr[1*3+d]*vy + Rr[2*3+d]*vz;   // R^T * v
                nrm += ol[d]*ol[d];
                f[128 + h*24 + p*3 + d] = f2bf(ol[d]);
            }
            f[224 + h*PV + p] = f2bf(sqrtf(nrm));
        }
    }
}

// ---------------- launch ----------------
static inline char* bump(char*& p, size_t bytes){
    char* r = p; p += (bytes + 255) & ~(size_t)255; return r;
}

extern "C" void kernel_launch(void* const* d_in, const int* in_sizes, int n_in,
                              void* d_out, int out_size, void* d_ws, size_t ws_size,
                              hipStream_t stream)
{
    const int*   seq       = (const int*)  d_in[0];
    const int*   seq_idx   = (const int*)  d_in[1];
    const int*   chain_idx = (const int*)  d_in[2];
    const float* mask      = (const float*)d_in[3];
    const float* rigid7    = (const float*)d_in[4];
    const float* t         = (const float*)d_in[5];
    const float* bonds     = (const float*)d_in[6];
    const float* W_time    = (const float*)d_in[7];
    const float* W_node_idx= (const float*)d_in[8];
    const float* W_seq     = (const float*)d_in[9];
    const float* W_relpos  = (const float*)d_in[10];
    const float* W_rbf     = (const float*)d_in[11];
    const float* W_chain   = (const float*)d_in[12];
    const float* W_bond    = (const float*)d_in[13];
    const float* Wq        = (const float*)d_in[14];
    const float* Wk        = (const float*)d_in[15];
    const float* Wv        = (const float*)d_in[16];
    const float* Wqp       = (const float*)d_in[17];
    const float* Wkp       = (const float*)d_in[18];
    const float* Wvp       = (const float*)d_in[19];
    const float* Wb        = (const float*)d_in[20];
    const float* gamma     = (const float*)d_in[21];
    const float* Wout      = (const float*)d_in[22];
    float* out = (float*)d_out;

    char* p = (char*)d_ws;
    float*  tvec    = (float*) bump(p, BB*CS*4);
    float*  WbRel   = (float*) bump(p, NREL*HH*4);
    float*  WbRbf   = (float*) bump(p, NRBF*HH*4);
    float*  WbCh    = (float*) bump(p, HH*4);
    float*  WbBd    = (float*) bump(p, HH*4);
    float*  cgam    = (float*) bump(p, HH*4);
    float*  node    = (float*) bump(p, (size_t)ROWS*CS*4);
    bf16_t* nodeBF  = (bf16_t*)bump(p, (size_t)ROWS*CS*2);
    float*  Rm      = (float*) bump(p, (size_t)ROWS*9*4);
    float*  trans   = (float*) bump(p, (size_t)ROWS*4*4);
    float*  lin     = (float*) bump(p, (size_t)ROWS*LINC*4);
    bf16_t* WcatT   = (bf16_t*)bump(p, (size_t)LINC*CS*2);
    bf16_t* WoutT   = (bf16_t*)bump(p, (size_t)CS*FEAT*2);
    bf16_t* Aq      = (bf16_t*)bump(p, (size_t)BB*HH*NN*KATT*2);
    bf16_t* Ak      = (bf16_t*)bump(p, (size_t)BB*HH*NN*KATT*2);
    float*  dk      = (float*) bump(p, (size_t)BB*HH*NN*4);
    bf16_t* vT      = (bf16_t*)bump(p, (size_t)BB*HH*CC*NN*2);
    bf16_t* vgT     = (bf16_t*)bump(p, (size_t)BB*HH*32*NN*2);
    float*  wlbias  = (float*) bump(p, (size_t)BB*NN*NN*HH*4);
    float*  logits  = (float*) bump(p, (size_t)BB*HH*NN*NN*4);
    bf16_t* probsBF = (bf16_t*)bump(p, (size_t)BB*HH*NN*NN*2);
    float*  obuf    = (float*) bump(p, (size_t)BB*HH*NN*32*4);
    float*  ogbuf   = (float*) bump(p, (size_t)BB*HH*NN*32*4);
    bf16_t* featsBF = (bf16_t*)bump(p, (size_t)ROWS*FEAT*2);
    (void)ws_size; (void)in_sizes; (void)n_in; (void)out_size;

    // 1) small precompute
    hipLaunchKernelGGL(k_prep, dim3(1), dim3(256), 0, stream,
                       t, gamma, W_time, W_relpos, W_rbf, W_chain, W_bond, Wb,
                       tvec, WbRel, WbRbf, WbCh, WbBd, cgam);
    // 2) pack weights -> bf16 K-major
    hipLaunchKernelGGL(k_pack_weights, dim3(512), dim3(256), 0, stream,
                       Wq, Wk, Wv, Wqp, Wkp, Wvp, Wout, WcatT, WoutT);
    // 3) node embedding + frames
    hipLaunchKernelGGL(k_node, dim3(ROWS), dim3(CS), 0, stream,
                       seq, seq_idx, rigid7, W_node_idx, W_seq, tvec, node, nodeBF, Rm, trans);
    // 4) node @ [Wq|Wk|Wv|Wq_pts|Wk_pts|Wv_pts]  (1024 x 384 x 576; 64x64 blocks, async-LDS)
    hipLaunchKernelGGL(k_wmma_gemm_blk, dim3(LINC/64, ROWS/64), dim3(128), 0, stream,
                       nodeBF, WcatT, lin, (const float*)nullptr, CS, CS, CS, LINC);
    // 5) pack score operands / vT / vgT
    hipLaunchKernelGGL(k_pack_qkv, dim3(32), dim3(256), 0, stream,
                       lin, Rm, trans, cgam, Aq, Ak, dk, vT, vgT);
    // 6) projected pair bias
    hipLaunchKernelGGL(k_pair, dim3(2048), dim3(256), 0, stream,
                       seq_idx, chain_idx, bonds, trans, WbRel, WbRbf, WbCh, WbBd, wlbias);
    // 7) attention logits (WMMA K=64: qk + point term fused)
    hipLaunchKernelGGL(k_scores, dim3(NN/16, NN/16, BB*HH), dim3(32), 0, stream,
                       Aq, Ak, wlbias, dk, mask, logits);
    // 8) softmax
    hipLaunchKernelGGL(k_softmax, dim3(BB*HH*NN), dim3(128), 0, stream, logits, probsBF);
    // 9) o = P @ v   (512 x 512 x 32 per (b,h), WMMA)
    hipLaunchKernelGGL(k_wmma_gemm, dim3(2, NN/16, BB*HH), dim3(32), 0, stream,
                       probsBF, vT, obuf, (const float*)nullptr,
                       NN, NN, NN, 32, (long long)NN*NN, (long long)CC*NN, (long long)NN*32);
    // 10) og = P @ vg (padded to 32 cols, WMMA)
    hipLaunchKernelGGL(k_wmma_gemm, dim3(2, NN/16, BB*HH), dim3(32), 0, stream,
                       probsBF, vgT, ogbuf, (const float*)nullptr,
                       NN, NN, NN, 32, (long long)NN*NN, (long long)32*NN, (long long)NN*32);
    // 11) opair via factored statistics
    hipLaunchKernelGGL(k_opair, dim3(ROWS), dim3(128), 0, stream,
                       probsBF, seq_idx, chain_idx, bonds, trans,
                       W_relpos, W_rbf, W_chain, W_bond, featsBF);
    // 12) o/ol/onorm into feats
    hipLaunchKernelGGL(k_finalize, dim3(32), dim3(256), 0, stream,
                       obuf, ogbuf, Rm, trans, featsBF);
    // 13) out = node + feats @ Wout  (1024 x 768 x 384; 64x64 blocks, async-LDS)
    hipLaunchKernelGGL(k_wmma_gemm_blk, dim3(CS/64, ROWS/64), dim3(128), 0, stream,
                       featsBF, WoutT, out, node, FEAT, FEAT, FEAT, CS);
}
